// GNN_84928683311960
// MI455X (gfx1250) — matrix operand order repaired and checked by reading, not compiled, not run on hardware
//
#include <hip/hip_runtime.h>

typedef __attribute__((ext_vector_type(16))) _Float16 v16h;
typedef __attribute__((ext_vector_type(8)))  _Float16 v8h;
typedef __attribute__((ext_vector_type(8)))  float    v8f;

#define N_NODES 50000
#define N_EDGES 1600000
#define HID     64
#define NGRAPH  16

__device__ __forceinline__ v8f wmma_f16(v16h a, v16h b, v8f c) {
    // D = A(16x32 f16) * B(32x16 f16) + C(16x16 f32)
    return __builtin_amdgcn_wmma_f32_16x16x32_f16(false, a, false, b,
                                                  (short)0, c, false, false);
}

// Build B fragment: W is KxN row-major f32 (stride 64), fragment covers
// rows [kbase, kbase+32), cols [t*16, t*16+16). Lane = col (lo), K split by hi.
__device__ __forceinline__ v16h build_bfrag(const float* __restrict__ W,
                                            int kbase, int col, int hi) {
    v16h b;
#pragma unroll
    for (int v = 0; v < 8; ++v) {
        int k0 = kbase + hi * 8 + 2 * v + (v >= 4 ? 8 : 0);
        b[2 * v]     = (_Float16)W[k0 * 64 + col];
        b[2 * v + 1] = (_Float16)W[(k0 + 1) * 64 + col];
    }
    return b;
}

// ---------------------------------------------------------------------------
// Zero workspace (agg + pooled)
// ---------------------------------------------------------------------------
__global__ void zero4_kernel(float4* __restrict__ p, int n4) {
    int i = blockIdx.x * blockDim.x + threadIdx.x;
    int stride = gridDim.x * blockDim.x;
    float4 z = make_float4(0.f, 0.f, 0.f, 0.f);
    for (; i < n4; i += stride) p[i] = z;
}

// ---------------------------------------------------------------------------
// Edge model: e = relu([x[row],x[col],ea] @ ew1 + eb1) @ ew2 + eb2
// fused scatter-add into agg[col]. One wave = 16 edges, WMMA for layer 2.
// ---------------------------------------------------------------------------
__global__ void __launch_bounds__(256) edge_kernel(
    const float* __restrict__ x, const int* __restrict__ ei,
    const float* __restrict__ ea,
    const float* __restrict__ ew1, const float* __restrict__ eb1,
    const float* __restrict__ ew2, const float* __restrict__ eb2,
    float* __restrict__ agg)
{
    __shared__ float s_w1[5 * 64];
    __shared__ float s_b1[64];
    __shared__ int   s_col[8][16];

    const int tid = threadIdx.x;
    for (int t = tid; t < 320; t += 256) s_w1[t] = ew1[t];
    if (tid < 64) s_b1[tid] = eb1[tid];
    __syncthreads();

    const int lane = tid & 31;
    const int wave = tid >> 5;
    const int lo   = lane & 15;
    const int hi   = lane >> 4;

    // Hoisted per-wave constants: ew2 B-fragments + eb2 bias
    v16h bf[4][2];
#pragma unroll
    for (int t = 0; t < 4; ++t) {
#pragma unroll
        for (int kt = 0; kt < 2; ++kt)
            bf[t][kt] = build_bfrag(ew2, kt * 32, t * 16 + lo, hi);
    }
    float bias2[4];
#pragma unroll
    for (int t = 0; t < 4; ++t) bias2[t] = eb2[t * 16 + lo];

    const int numTiles = N_EDGES / 16;
    for (int tile = blockIdx.x * 8 + wave; tile < numTiles; tile += gridDim.x * 8) {
        const int e  = tile * 16 + lo;
        const int er = ei[e];
        const int ec = ei[N_EDGES + e];
        float inp[5];
        inp[0] = x[er];
        inp[1] = x[ec];
        inp[2] = ea[e * 3 + 0];
        inp[3] = ea[e * 3 + 1];
        inp[4] = ea[e * 3 + 2];
        s_col[wave][lo] = ec;  // both half-waves write identical value

        // Layer 1 computed directly in A-fragment arrangement:
        // lane holds row (edge) lo; K(hidden col) groups g*16 + hi*8 .. +8
        v16h af[2];
#pragma unroll
        for (int g = 0; g < 4; ++g) {
            const int jb = g * 16 + hi * 8;
            float acc[8];
#pragma unroll
            for (int j = 0; j < 8; ++j) acc[j] = s_b1[jb + j];
#pragma unroll
            for (int k = 0; k < 5; ++k) {
                const float iv = inp[k];
#pragma unroll
                for (int j = 0; j < 8; ++j) acc[j] += iv * s_w1[k * 64 + jb + j];
            }
#pragma unroll
            for (int j = 0; j < 8; ++j) {
                float r = acc[j] > 0.f ? acc[j] : 0.f;
                af[g >> 1][(g & 1) * 8 + j] = (_Float16)r;
            }
        }

        asm volatile("" ::: "memory");  // keep s_col store before cross-lane reads
        // 32-bit element offsets (max 12.8M < 2^31) -> saddr+voffset+imm atomics
        unsigned voff[8];
#pragma unroll
        for (int r = 0; r < 8; ++r)
            voff[r] = (unsigned)s_col[wave][r + 8 * hi] * 64u + (unsigned)lo;

        // Layer 2: 4 N-tiles x (2 WMMA over K=64), then fused scatter-add
#pragma unroll
        for (int t = 0; t < 4; ++t) {
            v8f c;
#pragma unroll
            for (int r = 0; r < 8; ++r) c[r] = bias2[t];
            c = wmma_f16(af[0], bf[t][0], c);
            c = wmma_f16(af[1], bf[t][1], c);
#pragma unroll
            for (int r = 0; r < 8; ++r)
                atomicAdd(agg + voff[r] + t * 16, c[r]);  // imm disp = t*64 bytes
        }
    }
}

// ---------------------------------------------------------------------------
// Node model: h = relu([x,agg] @ nw1 + nb1) @ nw2 + nb2, pooled into LDS
// then one global flush per block. One wave = 16 nodes.
// ---------------------------------------------------------------------------
__global__ void __launch_bounds__(256) node_kernel(
    const float* __restrict__ x, const float* __restrict__ agg,
    const int* __restrict__ batch,
    const float* __restrict__ nw1, const float* __restrict__ nb1,
    const float* __restrict__ nw2, const float* __restrict__ nb2,
    float* __restrict__ pooled)
{
    __shared__ float s_pool[NGRAPH * 64];
    __shared__ float s_x[8][16];
    __shared__ int   s_g[8][16];
    __shared__ __align__(16) _Float16 s_h1[8][16][64];

    const int tid = threadIdx.x;
    for (int t = tid; t < NGRAPH * 64; t += 256) s_pool[t] = 0.f;
    __syncthreads();

    const int lane = tid & 31;
    const int wave = tid >> 5;
    const int lo   = lane & 15;
    const int hi   = lane >> 4;

    v16h bf1[4][2], bf2[4][2];
#pragma unroll
    for (int t = 0; t < 4; ++t) {
#pragma unroll
        for (int kt = 0; kt < 2; ++kt) {
            // nw1 rows 1..64 are the agg coefficients (row 0 is x)
            bf1[t][kt] = build_bfrag(nw1 + 64, kt * 32, t * 16 + lo, hi);
            bf2[t][kt] = build_bfrag(nw2,      kt * 32, t * 16 + lo, hi);
        }
    }
    float w0[4], b1v[4], b2v[4];
#pragma unroll
    for (int t = 0; t < 4; ++t) {
        const int col = t * 16 + lo;
        w0[t]  = nw1[col];   // nw1[0][col]
        b1v[t] = nb1[col];
        b2v[t] = nb2[col];
    }

    const int numTiles = N_NODES / 16;  // 3125
    for (int tile = blockIdx.x * 8 + wave; tile < numTiles; tile += gridDim.x * 8) {
        const int i = tile * 16 + lo;
        const unsigned rowoff = (unsigned)i * 64u + (unsigned)(hi * 8);

        // A-fragment of agg row i, straight from global (contiguous 32B chunks)
        v16h af[2];
#pragma unroll
        for (int g = 0; g < 4; ++g) {
            const float4* p = (const float4*)(agg + rowoff + g * 16);
            float4 u = p[0], w = p[1];
            af[g >> 1][(g & 1) * 8 + 0] = (_Float16)u.x;
            af[g >> 1][(g & 1) * 8 + 1] = (_Float16)u.y;
            af[g >> 1][(g & 1) * 8 + 2] = (_Float16)u.z;
            af[g >> 1][(g & 1) * 8 + 3] = (_Float16)u.w;
            af[g >> 1][(g & 1) * 8 + 4] = (_Float16)w.x;
            af[g >> 1][(g & 1) * 8 + 5] = (_Float16)w.y;
            af[g >> 1][(g & 1) * 8 + 6] = (_Float16)w.z;
            af[g >> 1][(g & 1) * 8 + 7] = (_Float16)w.w;
        }
        s_x[wave][lo] = x[i];
        s_g[wave][lo] = batch[i];
        asm volatile("" ::: "memory");
        float xv[8]; int gg[8];
#pragma unroll
        for (int r = 0; r < 8; ++r) {
            xv[r] = s_x[wave][r + 8 * hi];
            gg[r] = s_g[wave][r + 8 * hi];
        }

        // Layer 1: C preloaded with bias + x-term, WMMA over agg; relu -> LDS (f16)
#pragma unroll
        for (int t = 0; t < 4; ++t) {
            v8f c;
#pragma unroll
            for (int r = 0; r < 8; ++r) c[r] = b1v[t] + xv[r] * w0[t];
            c = wmma_f16(af[0], bf1[t][0], c);
            c = wmma_f16(af[1], bf1[t][1], c);
#pragma unroll
            for (int r = 0; r < 8; ++r) {
                float rl = c[r] > 0.f ? c[r] : 0.f;
                s_h1[wave][r + 8 * hi][t * 16 + lo] = (_Float16)rl;
            }
        }
        asm volatile("" ::: "memory");  // wave-local LDS is in-order; block reordering

        // Re-fragment h1 (layout transpose via LDS)
        v16h hf[2];
#pragma unroll
        for (int g = 0; g < 4; ++g) {
            v8h hv = *(const v8h*)&s_h1[wave][lo][g * 16 + hi * 8];
#pragma unroll
            for (int j = 0; j < 8; ++j)
                hf[g >> 1][(g & 1) * 8 + j] = hv[j];
        }

        // Layer 2 + pooled accumulation in LDS (ds_add_f32)
#pragma unroll
        for (int t = 0; t < 4; ++t) {
            v8f c;
#pragma unroll
            for (int r = 0; r < 8; ++r) c[r] = b2v[t];
            c = wmma_f16(hf[0], bf2[t][0], c);
            c = wmma_f16(hf[1], bf2[t][1], c);
#pragma unroll
            for (int r = 0; r < 8; ++r)
                atomicAdd(&s_pool[gg[r] * 64 + t * 16 + lo], c[r]);
        }
    }

    __syncthreads();
    for (int t = tid; t < NGRAPH * 64; t += 256) atomicAdd(&pooled[t], s_pool[t]);
}

// ---------------------------------------------------------------------------
// Head: 3x (16x64 @ 64x64 + relu) via WMMA chain, final 64->2 scalar. 1 wave.
// ---------------------------------------------------------------------------
__global__ void __launch_bounds__(32) head_kernel(
    const float* __restrict__ pooled,
    const float* __restrict__ ow1, const float* __restrict__ ob1,
    const float* __restrict__ ow2, const float* __restrict__ ob2,
    const float* __restrict__ ow3, const float* __restrict__ ob3,
    const float* __restrict__ ow4, const float* __restrict__ ob4,
    float* __restrict__ out)
{
    __shared__ __align__(16) _Float16 s_t[16][64];
    __shared__ float s_f[16][64];

    const int lane = threadIdx.x & 31;
    const int lo   = lane & 15;
    const int hi   = lane >> 4;

    v16h af[2];
#pragma unroll
    for (int g = 0; g < 4; ++g) {
        const float4* p = (const float4*)&pooled[lo * 64 + g * 16 + hi * 8];
        float4 u = p[0], w = p[1];
        af[g >> 1][(g & 1) * 8 + 0] = (_Float16)u.x;
        af[g >> 1][(g & 1) * 8 + 1] = (_Float16)u.y;
        af[g >> 1][(g & 1) * 8 + 2] = (_Float16)u.z;
        af[g >> 1][(g & 1) * 8 + 3] = (_Float16)u.w;
        af[g >> 1][(g & 1) * 8 + 4] = (_Float16)w.x;
        af[g >> 1][(g & 1) * 8 + 5] = (_Float16)w.y;
        af[g >> 1][(g & 1) * 8 + 6] = (_Float16)w.z;
        af[g >> 1][(g & 1) * 8 + 7] = (_Float16)w.w;
    }

    const float* Ws[3] = {ow1, ow2, ow3};
    const float* Bs[3] = {ob1, ob2, ob3};
    for (int L = 0; L < 3; ++L) {
        const float* W  = Ws[L];
        const float* Bv = Bs[L];
        v16h bf[4][2];
#pragma unroll
        for (int t = 0; t < 4; ++t)
#pragma unroll
            for (int kt = 0; kt < 2; ++kt)
                bf[t][kt] = build_bfrag(W, kt * 32, t * 16 + lo, hi);

#pragma unroll
        for (int t = 0; t < 4; ++t) {
            float bias = Bv[t * 16 + lo];
            v8f c;
#pragma unroll
            for (int r = 0; r < 8; ++r) c[r] = bias;
            c = wmma_f16(af[0], bf[t][0], c);
            c = wmma_f16(af[1], bf[t][1], c);
#pragma unroll
            for (int r = 0; r < 8; ++r) {
                float rl = c[r] > 0.f ? c[r] : 0.f;
                s_t[r + 8 * hi][t * 16 + lo] = (_Float16)rl;
                if (L == 2) s_f[r + 8 * hi][t * 16 + lo] = rl;
            }
        }
        asm volatile("" ::: "memory");
#pragma unroll
        for (int g = 0; g < 4; ++g) {
            v8h hv = *(const v8h*)&s_t[lo][g * 16 + hi * 8];
#pragma unroll
            for (int j = 0; j < 8; ++j)
                af[g >> 1][(g & 1) * 8 + j] = hv[j];
        }
        asm volatile("" ::: "memory");
    }

    // Final Linear(64,2): lane = (graph lo, out-col hi), f32 from LDS
    const int m = lo, cc = hi;
    float acc = ob4[cc];
#pragma unroll 8
    for (int k = 0; k < 64; ++k) acc += s_f[m][k] * ow4[k * 2 + cc];
    out[m * 2 + cc] = acc;
}

// ---------------------------------------------------------------------------
extern "C" void kernel_launch(void* const* d_in, const int* in_sizes, int n_in,
                              void* d_out, int out_size, void* d_ws, size_t ws_size,
                              hipStream_t stream) {
    (void)in_sizes; (void)n_in; (void)out_size; (void)ws_size;
    const float* x    = (const float*)d_in[0];
    const int*   ei   = (const int*)d_in[1];
    const float* ea   = (const float*)d_in[2];
    const int*   batch= (const int*)d_in[3];
    const float* ew1  = (const float*)d_in[4];
    const float* eb1  = (const float*)d_in[5];
    const float* ew2  = (const float*)d_in[6];
    const float* eb2  = (const float*)d_in[7];
    const float* nw1  = (const float*)d_in[8];
    const float* nb1  = (const float*)d_in[9];
    const float* nw2  = (const float*)d_in[10];
    const float* nb2  = (const float*)d_in[11];
    const float* ow1  = (const float*)d_in[12];
    const float* ob1  = (const float*)d_in[13];
    const float* ow2  = (const float*)d_in[14];
    const float* ob2  = (const float*)d_in[15];
    const float* ow3  = (const float*)d_in[16];
    const float* ob3  = (const float*)d_in[17];
    const float* ow4  = (const float*)d_in[18];
    const float* ob4  = (const float*)d_in[19];

    float* agg    = (float*)d_ws;                       // N*64 f32
    float* pooled = agg + (size_t)N_NODES * 64;         // G*64 f32

    const int zn4 = (N_NODES * 64 + NGRAPH * 64) / 4;
    zero4_kernel<<<1024, 256, 0, stream>>>((float4*)d_ws, zn4);
    edge_kernel<<<1024, 256, 0, stream>>>(x, ei, ea, ew1, eb1, ew2, eb2, agg);
    node_kernel<<<256, 256, 0, stream>>>(x, agg, batch, nw1, nb1, nw2, nb2, pooled);
    head_kernel<<<1, 32, 0, stream>>>(pooled, ow1, ob1, ow2, ob2, ow3, ob3,
                                      ow4, ob4, (float*)d_out);
}
